// Decoder_87754771792060
// MI455X (gfx1250) — matrix operand order repaired and checked by reading, not compiled, not run on hardware
//
#include <hip/hip_runtime.h>
#include <hip/hip_bf16.h>
#include <stdint.h>

#define BB 16
#define TT 32
#define SS 400
#define EE 128
#define HH 512
#define VV 32000
#define OOVV 50
#define VX (VV + OOVV)

typedef __bf16 bf16;
typedef __attribute__((ext_vector_type(16))) bf16 v16bf;
typedef __attribute__((ext_vector_type(8))) float v8f;

__device__ __forceinline__ bf16 tobf(float f) { return (bf16)f; }

// Fast transcendentals on the native v_exp_f32 pipe.
__device__ __forceinline__ float fast_sigmoid(float x) { return 1.f / (1.f + __expf(-x)); }
__device__ __forceinline__ float fast_tanh(float x) { return 1.f - 2.f / (__expf(2.f * x) + 1.f); }

// A fragment: 16x32 bf16 tile, rows m0..m0+15 (one per lane%16), K per ISA layout.
// Per lane: K = {k0 + half*8 + 0..7} then {k0 + 16 + half*8 + 0..7} -> two 16B loads.
__device__ __forceinline__ v16bf load_a_frag(const bf16* A, int lda, int m0, int k0, int lane) {
  int m = m0 + (lane & 15);
  int half = lane >> 4;
  const bf16* p0 = A + (size_t)m * lda + k0 + half * 8;
  v16bf r;
  ((uint4*)&r)[0] = *(const uint4*)(p0);
  ((uint4*)&r)[1] = *(const uint4*)(p0 + 16);
  return r;
}

// Same A layout but f32 source, converted in-register (one-time enc_proj GEMM).
__device__ __forceinline__ v16bf load_a_frag_f32(const float* A, int lda, int m0, int k0, int lane) {
  int m = m0 + (lane & 15);
  int half = lane >> 4;
  const float* p0 = A + (size_t)m * lda + k0 + half * 8;
  const float* p1 = p0 + 16;
  v16bf r;
#pragma unroll
  for (int i = 0; i < 8; ++i) { r[i] = tobf(p0[i]); r[i + 8] = tobf(p1[i]); }
  return r;
}

// B fragment: 32x16 bf16 tile of W^T where W is row-major (N x K).
// Per lane (col n = lane%16): K = {k0 + half*8 + 0..7} then {k0 + 16 + half*8 + 0..7} -> two 16B loads.
__device__ __forceinline__ v16bf load_b_frag(const bf16* W, int ldw, int n0, int k0, int lane) {
  int n = n0 + (lane & 15);
  int half = lane >> 4;
  const bf16* p = W + (size_t)n * ldw + k0 + half * 8;
  v16bf r;
  ((uint4*)&r)[0] = *(const uint4*)(p);
  ((uint4*)&r)[1] = *(const uint4*)(p + 16);
  return r;
}

__device__ __forceinline__ v8f wmma_bf16(v16bf a, v16bf b, v8f c) {
  return __builtin_amdgcn_wmma_f32_16x16x32_bf16(false, a, false, b, (short)0, c, false, false);
}

// ---------------- setup kernels ----------------

__global__ void k_f32_to_bf16(const float* __restrict__ src, bf16* __restrict__ dst, int n) {
  int i = blockIdx.x * blockDim.x + threadIdx.x;
  if (i < n) dst[i] = tobf(src[i]);
}

__global__ void k_split_energy(const float* __restrict__ W_energy, bf16* __restrict__ Wd_bf,
                               bf16* __restrict__ We_bf) {
  int i = blockIdx.x * blockDim.x + threadIdx.x;
  if (i < HH * HH) {
    int r = i / HH, c = i % HH;
    Wd_bf[i] = tobf(W_energy[(size_t)r * 2 * HH + c]);
    We_bf[i] = tobf(W_energy[(size_t)r * 2 * HH + HH + c]);
  }
}

__global__ void k_init_state(const float* __restrict__ h0, const float* __restrict__ c0,
                             float* h, float* c, bf16* h_bf, float* ctx, bf16* ctx_bf,
                             float* cov) {
  int i = blockIdx.x * blockDim.x + threadIdx.x;
  if (i < BB * HH) {
    h[i] = h0[i];
    c[i] = c0[i];
    h_bf[i] = tobf(h0[i]);
    ctx[i] = 0.f;
    ctx_bf[i] = tobf(0.f);
  }
  if (i < BB * SS) cov[i] = 0.f;
}

// enc_proj[b,s,k] = sum_h enc[b,s,h]*W_e[k,h] + b_energy[k]; M=6400,N=512,K=512
// 2 N-tiles per wave sharing the (expensive, f32->bf16 converted) A fragment.
__global__ void __launch_bounds__(128) k_encproj(const float* __restrict__ enc,
                                                 const bf16* __restrict__ We_bf,
                                                 const float* __restrict__ b_energy,
                                                 float* __restrict__ enc_proj) {
  int lane = threadIdx.x & 31, wave = threadIdx.x >> 5;
  int id = blockIdx.x * 4 + wave;        // 6400 wave-jobs = 400 (M) x 16 (N pairs)
  int m0 = (id >> 4) * 16, n0 = (id & 15) * 32;
  v8f acc0 = {}, acc1 = {};
  for (int k0 = 0; k0 < HH; k0 += 32) {
    v16bf a = load_a_frag_f32(enc, HH, m0, k0, lane);
    acc0 = wmma_bf16(a, load_b_frag(We_bf, HH, n0, k0, lane), acc0);
    acc1 = wmma_bf16(a, load_b_frag(We_bf, HH, n0 + 16, k0, lane), acc1);
  }
  int half = lane >> 4;
#pragma unroll
  for (int j = 0; j < 2; ++j) {
    int n = n0 + j * 16 + (lane & 15);
    float bias = b_energy[n];
    v8f acc = j ? acc1 : acc0;
#pragma unroll
    for (int r = 0; r < 8; ++r)
      enc_proj[(size_t)(m0 + r + 8 * half) * HH + n] = acc[r] + bias;
  }
}

// ---------------- per-step kernels ----------------

// x = concat(emb_t, ctx) @ W_xctx^T + b_xctx ; M=16,N=128,K=640 ; 1 block, 8 waves
__global__ void __launch_bounds__(256) k_x(const bf16* __restrict__ emb_bf, int t,
                                           const bf16* __restrict__ ctx_bf,
                                           const bf16* __restrict__ Wx_bf,
                                           const float* __restrict__ b_xctx,
                                           float* __restrict__ x, bf16* __restrict__ x_bf) {
  int lane = threadIdx.x & 31, wave = threadIdx.x >> 5;
  int n0 = wave * 16;
  v8f acc = {};
  for (int k0 = 0; k0 < EE + HH; k0 += 32) {
    v16bf a = (k0 < EE) ? load_a_frag(emb_bf + (size_t)t * EE, TT * EE, 0, k0, lane)
                        : load_a_frag(ctx_bf, HH, 0, k0 - EE, lane);
    acc = wmma_bf16(a, load_b_frag(Wx_bf, EE + HH, n0, k0, lane), acc);
  }
  int n = n0 + (lane & 15), half = lane >> 4;
  float bias = b_xctx[n];
#pragma unroll
  for (int r = 0; r < 8; ++r) {
    int m = r + 8 * half;
    float v = acc[r] + bias;
    x[m * EE + n] = v;
    x_bf[m * EE + n] = tobf(v);
  }
}

// gates = concat(x,h) @ [W_ih|W_hh]^T + b ; M=16,N=2048,K=640 ; 2 tiles/wave, 16 blocks
__global__ void __launch_bounds__(128) k_gates(const bf16* __restrict__ x_bf,
                                               const bf16* __restrict__ h_bf,
                                               const bf16* __restrict__ Wih_bf,
                                               const bf16* __restrict__ Whh_bf,
                                               const float* __restrict__ b_ih,
                                               const float* __restrict__ b_hh,
                                               float* __restrict__ gates) {
  int lane = threadIdx.x & 31, wave = threadIdx.x >> 5;
  int n0 = (blockIdx.x * 4 + wave) * 32;  // 64 wave-jobs x 32 cols
  v8f acc0 = {}, acc1 = {};
  for (int k0 = 0; k0 < EE + HH; k0 += 32) {
    v16bf a;
    const bf16* W;
    int ldw, lk;
    if (k0 < EE) {
      a = load_a_frag(x_bf, EE, 0, k0, lane);
      W = Wih_bf; ldw = EE; lk = k0;
    } else {
      a = load_a_frag(h_bf, HH, 0, k0 - EE, lane);
      W = Whh_bf; ldw = HH; lk = k0 - EE;
    }
    acc0 = wmma_bf16(a, load_b_frag(W, ldw, n0, lk, lane), acc0);
    acc1 = wmma_bf16(a, load_b_frag(W, ldw, n0 + 16, lk, lane), acc1);
  }
  int half = lane >> 4;
#pragma unroll
  for (int j = 0; j < 2; ++j) {
    int n = n0 + j * 16 + (lane & 15);
    float bias = b_ih[n] + b_hh[n];
    v8f acc = j ? acc1 : acc0;
#pragma unroll
    for (int r = 0; r < 8; ++r)
      gates[(size_t)(r + 8 * half) * (4 * HH) + n] = acc[r] + bias;
  }
}

__global__ void k_lstm(const float* __restrict__ gates, float* c, float* h, bf16* h_bf) {
  int i = blockIdx.x * blockDim.x + threadIdx.x;
  if (i >= BB * HH) return;
  int b = i / HH, j = i % HH;
  const float* g = gates + (size_t)b * 4 * HH;
  float ig = fast_sigmoid(g[j]);
  float fg = fast_sigmoid(g[HH + j]);
  float gg = fast_tanh(g[2 * HH + j]);
  float og = fast_sigmoid(g[3 * HH + j]);
  float cn = fg * c[i] + ig * gg;
  float hn = og * fast_tanh(cn);
  c[i] = cn;
  h[i] = hn;
  h_bf[i] = tobf(hn);
}

// dec_proj = h @ W_d^T ; M=16,N=512,K=512 ; 2 tiles/wave, 4 blocks
__global__ void __launch_bounds__(128) k_dproj(const bf16* __restrict__ h_bf,
                                               const bf16* __restrict__ Wd_bf,
                                               float* __restrict__ dproj) {
  int lane = threadIdx.x & 31, wave = threadIdx.x >> 5;
  int n0 = (blockIdx.x * 4 + wave) * 32;
  v8f acc0 = {}, acc1 = {};
  for (int k0 = 0; k0 < HH; k0 += 32) {
    v16bf a = load_a_frag(h_bf, HH, 0, k0, lane);
    acc0 = wmma_bf16(a, load_b_frag(Wd_bf, HH, n0, k0, lane), acc0);
    acc1 = wmma_bf16(a, load_b_frag(Wd_bf, HH, n0 + 16, k0, lane), acc1);
  }
  int half = lane >> 4;
#pragma unroll
  for (int j = 0; j < 2; ++j) {
    int n = n0 + j * 16 + (lane & 15);
    v8f acc = j ? acc1 : acc0;
#pragma unroll
    for (int r = 0; r < 8; ++r)
      dproj[(size_t)(r + 8 * half) * HH + n] = acc[r];
  }
}

// score[b,s] = v_attn . tanh(dproj[b]+enc_proj[b,s]+cov[b,s]*w_cov) ; one wave per (b,s)
__global__ void __launch_bounds__(256) k_score(const float* __restrict__ dproj,
                                               const float* __restrict__ enc_proj,
                                               const float* __restrict__ cov,
                                               const float* __restrict__ w_cov,
                                               const float* __restrict__ v_attn,
                                               float* __restrict__ score) {
  int lane = threadIdx.x & 31, wave = threadIdx.x >> 5;
  int p = blockIdx.x * 8 + wave;  // 6400 pairs
  int b = p / SS;
  float cv = cov[p];
  const float* ep = enc_proj + (size_t)p * HH;
  const float* dp = dproj + (size_t)b * HH;
  float sum = 0.f;
  for (int j = lane; j < HH; j += 32)
    sum += v_attn[j] * fast_tanh(dp[j] + ep[j] + cv * w_cov[j]);
#pragma unroll
  for (int o = 16; o > 0; o >>= 1) sum += __shfl_xor(sum, o, 32);
  if (lane == 0) score[p] = sum;
}

// softmax over S, mask+renorm, ctx = probs@enc, cov += probs, p_gen ; one block per b
__global__ void __launch_bounds__(512) k_attn_softmax(
    const float* __restrict__ score, const float* __restrict__ mask,
    const float* __restrict__ enc, const float* __restrict__ h, const float* __restrict__ x,
    const float* __restrict__ Wp, const float* __restrict__ bp, float* __restrict__ probs,
    float* __restrict__ cov, float* __restrict__ ctx, bf16* __restrict__ ctx_bf,
    float* __restrict__ pgen) {
  int b = blockIdx.x, tid = threadIdx.x;  // 512 threads
  __shared__ float sh[512];
  __shared__ float sp[SS];
  float sc = (tid < SS) ? score[b * SS + tid] : -1e30f;
  sh[tid] = sc;
  __syncthreads();
  for (int o = 256; o > 0; o >>= 1) {
    if (tid < o) sh[tid] = fmaxf(sh[tid], sh[tid + o]);
    __syncthreads();
  }
  float mx = sh[0];
  __syncthreads();
  float e = (tid < SS) ? __expf(sc - mx) : 0.f;
  sh[tid] = e;
  __syncthreads();
  for (int o = 256; o > 0; o >>= 1) {
    if (tid < o) sh[tid] += sh[tid + o];
    __syncthreads();
  }
  float se = sh[0];
  __syncthreads();
  float p1 = (tid < SS) ? (e / se) * mask[b * SS + tid] : 0.f;
  sh[tid] = p1;
  __syncthreads();
  for (int o = 256; o > 0; o >>= 1) {
    if (tid < o) sh[tid] += sh[tid + o];
    __syncthreads();
  }
  float sm = sh[0];
  __syncthreads();
  float p = p1 / (sm + 1e-12f);
  if (tid < SS) {
    sp[tid] = p;
    probs[b * SS + tid] = p;
    cov[b * SS + tid] += p;
  }
  __syncthreads();
  float acc = 0.f;
  const float* eb = enc + (size_t)b * SS * HH + tid;
  for (int s = 0; s < SS; ++s) acc += sp[s] * eb[(size_t)s * HH];
  ctx[b * HH + tid] = acc;
  ctx_bf[b * HH + tid] = tobf(acc);
  float contrib = acc * Wp[tid] + h[b * HH + tid] * Wp[HH + tid];
  if (tid < EE) contrib += x[b * EE + tid] * Wp[2 * HH + tid];
  sh[tid] = contrib;
  __syncthreads();
  for (int o = 256; o > 0; o >>= 1) {
    if (tid < o) sh[tid] += sh[tid + o];
    __syncthreads();
  }
  if (tid == 0) pgen[b] = fast_sigmoid(sh[0] + bp[0]);
}

// hh = concat(h, ctx) @ W_ad^T + b_ad ; M=16,N=512,K=1024 ; 2 tiles/wave, 4 blocks
__global__ void __launch_bounds__(128) k_hh(const bf16* __restrict__ h_bf,
                                            const bf16* __restrict__ ctx_bf,
                                            const bf16* __restrict__ Wad_bf,
                                            const float* __restrict__ b_ad,
                                            bf16* __restrict__ hh_bf) {
  int lane = threadIdx.x & 31, wave = threadIdx.x >> 5;
  int n0 = (blockIdx.x * 4 + wave) * 32;
  v8f acc0 = {}, acc1 = {};
  for (int k0 = 0; k0 < 2 * HH; k0 += 32) {
    v16bf a = (k0 < HH) ? load_a_frag(h_bf, HH, 0, k0, lane)
                        : load_a_frag(ctx_bf, HH, 0, k0 - HH, lane);
    acc0 = wmma_bf16(a, load_b_frag(Wad_bf, 2 * HH, n0, k0, lane), acc0);
    acc1 = wmma_bf16(a, load_b_frag(Wad_bf, 2 * HH, n0 + 16, k0, lane), acc1);
  }
  int half = lane >> 4;
#pragma unroll
  for (int j = 0; j < 2; ++j) {
    int n = n0 + j * 16 + (lane & 15);
    float bias = b_ad[n];
    v8f acc = j ? acc1 : acc0;
#pragma unroll
    for (int r = 0; r < 8; ++r)
      hh_bf[(size_t)(r + 8 * half) * HH + n] = tobf(acc[r] + bias);
  }
}

// logits = hh @ W_vocab^T + b_vocab ; M=16,N=32000,K=512
// 4 N-tiles per wave: one shared A fragment feeds 4 independent WMMA chains.
__global__ void __launch_bounds__(128) k_vocab(const bf16* __restrict__ hh_bf,
                                               const bf16* __restrict__ Wv_bf,
                                               const float* __restrict__ b_vocab,
                                               float* __restrict__ logits) {
  int lane = threadIdx.x & 31, wave = threadIdx.x >> 5;
  int n0 = (blockIdx.x * 4 + wave) * 64;  // 500 wave-jobs x 64 cols
  v8f acc0 = {}, acc1 = {}, acc2 = {}, acc3 = {};
  for (int k0 = 0; k0 < HH; k0 += 32) {
    v16bf a = load_a_frag(hh_bf, HH, 0, k0, lane);
    acc0 = wmma_bf16(a, load_b_frag(Wv_bf, HH, n0, k0, lane), acc0);
    acc1 = wmma_bf16(a, load_b_frag(Wv_bf, HH, n0 + 16, k0, lane), acc1);
    acc2 = wmma_bf16(a, load_b_frag(Wv_bf, HH, n0 + 32, k0, lane), acc2);
    acc3 = wmma_bf16(a, load_b_frag(Wv_bf, HH, n0 + 48, k0, lane), acc3);
  }
  int half = lane >> 4;
#pragma unroll
  for (int j = 0; j < 4; ++j) {
    int n = n0 + j * 16 + (lane & 15);
    float bias = b_vocab[n];
    v8f acc = (j == 0) ? acc0 : (j == 1) ? acc1 : (j == 2) ? acc2 : acc3;
#pragma unroll
    for (int r = 0; r < 8; ++r)
      logits[(size_t)(r + 8 * half) * VV + n] = acc[r] + bias;
  }
}

__global__ void k_rowstats(const float* __restrict__ logits, float* rmax, float* rsum) {
  __shared__ float sh[256];
  int b = blockIdx.x, tid = threadIdx.x;
  const float* row = logits + (size_t)b * VV;
  float m = -1e30f;
  for (int v = tid; v < VV; v += 256) m = fmaxf(m, row[v]);
  sh[tid] = m;
  __syncthreads();
  for (int o = 128; o > 0; o >>= 1) {
    if (tid < o) sh[tid] = fmaxf(sh[tid], sh[tid + o]);
    __syncthreads();
  }
  float mx = sh[0];
  __syncthreads();
  float s = 0.f;
  for (int v = tid; v < VV; v += 256) s += __expf(row[v] - mx);
  sh[tid] = s;
  __syncthreads();
  for (int o = 128; o > 0; o >>= 1) {
    if (tid < o) sh[tid] += sh[tid + o];
    __syncthreads();
  }
  if (tid == 0) {
    rmax[b] = mx;
    rsum[b] = sh[0];
  }
}

__global__ void k_final(const float* __restrict__ logits, const float* __restrict__ rmax,
                        const float* __restrict__ rsum, const float* __restrict__ pgen,
                        const float* __restrict__ extraz, int t, float* __restrict__ out) {
  size_t i = (size_t)blockIdx.x * blockDim.x + threadIdx.x;
  if (i >= (size_t)BB * VX) return;
  int b = (int)(i / VX), v = (int)(i % VX);
  float val;
  if (v < VV)
    val = pgen[b] * __expf(logits[(size_t)b * VV + v] - rmax[b]) / rsum[b];
  else
    val = extraz[b * OOVV + (v - VV)];
  out[((size_t)b * TT + t) * VX + v] = val;
}

__global__ void k_scatter(const float* __restrict__ probs, const float* __restrict__ pgen,
                          const int* __restrict__ idx, int t, float* __restrict__ out) {
  int i = blockIdx.x * blockDim.x + threadIdx.x;
  if (i >= BB * SS) return;
  int b = i / SS;
  float a = (1.f - pgen[b]) * probs[i];
  atomicAdd(out + ((size_t)b * TT + t) * VX + idx[i], a);
}

// ---------------- host ----------------

extern "C" void kernel_launch(void* const* d_in, const int* in_sizes, int n_in, void* d_out,
                              int out_size, void* d_ws, size_t ws_size, hipStream_t stream) {
  (void)in_sizes; (void)n_in; (void)out_size; (void)ws_size;
  const float* emb      = (const float*)d_in[0];
  const float* h0       = (const float*)d_in[1];
  const float* c0       = (const float*)d_in[2];
  const float* enc      = (const float*)d_in[3];
  const float* mask     = (const float*)d_in[4];
  const float* extraz   = (const float*)d_in[5];
  const int*   extidx   = (const int*)d_in[6];
  const float* W_ih     = (const float*)d_in[7];
  const float* W_hh     = (const float*)d_in[8];
  const float* b_ih     = (const float*)d_in[9];
  const float* b_hh     = (const float*)d_in[10];
  const float* W_vocab  = (const float*)d_in[11];
  const float* b_vocab  = (const float*)d_in[12];
  const float* W_cov    = (const float*)d_in[13];
  const float* W_energy = (const float*)d_in[14];
  const float* b_energy = (const float*)d_in[15];
  const float* v_attn   = (const float*)d_in[16];
  const float* W_xctx   = (const float*)d_in[17];
  const float* b_xctx   = (const float*)d_in[18];
  const float* W_ad     = (const float*)d_in[19];
  const float* b_ad     = (const float*)d_in[20];
  const float* W_pgen   = (const float*)d_in[21];
  const float* b_pgen   = (const float*)d_in[22];
  float* out = (float*)d_out;

  char* ws = (char*)d_ws;
  size_t off = 0;
  auto carve = [&](size_t bytes) -> void* {
    void* p = ws + off;
    off = (off + bytes + 255) & ~(size_t)255;
    return p;
  };

  bf16* emb_bf = (bf16*)carve((size_t)BB * TT * EE * 2);
  bf16* Wx_bf  = (bf16*)carve((size_t)EE * (EE + HH) * 2);
  bf16* Wih_bf = (bf16*)carve((size_t)4 * HH * EE * 2);
  bf16* Whh_bf = (bf16*)carve((size_t)4 * HH * HH * 2);
  bf16* Wd_bf  = (bf16*)carve((size_t)HH * HH * 2);
  bf16* We_bf  = (bf16*)carve((size_t)HH * HH * 2);
  bf16* Wad_bf = (bf16*)carve((size_t)HH * 2 * HH * 2);
  bf16* Wv_bf  = (bf16*)carve((size_t)VV * HH * 2);
  float* enc_proj = (float*)carve((size_t)BB * SS * HH * 4);
  float* hbuf   = (float*)carve((size_t)BB * HH * 4);
  float* cbuf   = (float*)carve((size_t)BB * HH * 4);
  float* ctxbuf = (float*)carve((size_t)BB * HH * 4);
  float* dproj  = (float*)carve((size_t)BB * HH * 4);
  float* xbuf   = (float*)carve((size_t)BB * EE * 4);
  float* gates  = (float*)carve((size_t)BB * 4 * HH * 4);
  float* cov    = (float*)carve((size_t)BB * SS * 4);
  float* probs  = (float*)carve((size_t)BB * SS * 4);
  float* score  = (float*)carve((size_t)BB * SS * 4);
  float* logits = (float*)carve((size_t)BB * VV * 4);
  bf16* h_bf   = (bf16*)carve((size_t)BB * HH * 2);
  bf16* ctx_bf = (bf16*)carve((size_t)BB * HH * 2);
  bf16* x_bf   = (bf16*)carve((size_t)BB * EE * 2);
  bf16* hh_bf  = (bf16*)carve((size_t)BB * HH * 2);
  float* rmax = (float*)carve(BB * 4);
  float* rsum = (float*)carve(BB * 4);
  float* pg   = (float*)carve(BB * 4);

  // ---- one-time setup: bf16 weight conversion + enc_proj precompute ----
  k_f32_to_bf16<<<(BB * TT * EE + 255) / 256, 256, 0, stream>>>(emb, emb_bf, BB * TT * EE);
  k_f32_to_bf16<<<(EE * (EE + HH) + 255) / 256, 256, 0, stream>>>(W_xctx, Wx_bf, EE * (EE + HH));
  k_f32_to_bf16<<<(4 * HH * EE + 255) / 256, 256, 0, stream>>>(W_ih, Wih_bf, 4 * HH * EE);
  k_f32_to_bf16<<<(4 * HH * HH + 255) / 256, 256, 0, stream>>>(W_hh, Whh_bf, 4 * HH * HH);
  k_f32_to_bf16<<<(HH * 2 * HH + 255) / 256, 256, 0, stream>>>(W_ad, Wad_bf, HH * 2 * HH);
  k_f32_to_bf16<<<(VV * HH + 255) / 256, 256, 0, stream>>>(W_vocab, Wv_bf, VV * HH);
  k_split_energy<<<(HH * HH + 255) / 256, 256, 0, stream>>>(W_energy, Wd_bf, We_bf);
  k_init_state<<<(BB * HH + 255) / 256, 256, 0, stream>>>(h0, c0, hbuf, cbuf, h_bf, ctxbuf,
                                                          ctx_bf, cov);
  k_encproj<<<1600, 128, 0, stream>>>(enc, We_bf, b_energy, enc_proj);

  // ---- recurrent scan over T steps ----
  for (int t = 0; t < TT; ++t) {
    k_x<<<1, 256, 0, stream>>>(emb_bf, t, ctx_bf, Wx_bf, b_xctx, xbuf, x_bf);
    k_gates<<<16, 128, 0, stream>>>(x_bf, h_bf, Wih_bf, Whh_bf, b_ih, b_hh, gates);
    k_lstm<<<(BB * HH + 255) / 256, 256, 0, stream>>>(gates, cbuf, hbuf, h_bf);
    k_dproj<<<4, 128, 0, stream>>>(h_bf, Wd_bf, dproj);
    k_score<<<800, 256, 0, stream>>>(dproj, enc_proj, cov, W_cov, v_attn, score);
    k_attn_softmax<<<BB, 512, 0, stream>>>(score, mask, enc, hbuf, xbuf, W_pgen, b_pgen, probs,
                                           cov, ctxbuf, ctx_bf, pg);
    k_hh<<<4, 128, 0, stream>>>(h_bf, ctx_bf, Wad_bf, b_ad, hh_bf);
    k_vocab<<<125, 128, 0, stream>>>(hh_bf, Wv_bf, b_vocab, logits);
    k_rowstats<<<BB, 256, 0, stream>>>(logits, rmax, rsum);
    k_final<<<(BB * VX + 255) / 256, 256, 0, stream>>>(logits, rmax, rsum, pg, extraz, t, out);
    k_scatter<<<(BB * SS + 255) / 256, 256, 0, stream>>>(probs, pg, extidx, t, out);
  }
}
